// WassersteinBarycenter_31353261261219
// MI455X (gfx1250) — matrix operand order repaired and checked by reading, not compiled, not run on hardware
//
#include <hip/hip_runtime.h>
#include <hip/hip_bf16.h>

// Wasserstein barycenter (iterative Bregman projection), exp-domain
// reformulation:  the two logsumexp contractions are matvecs with
// K_mat = exp(-costs/eps) and K_mat^T, executed on the CDNA5 matrix
// engine via V_WMMA_F32_16X16X32_BF16.  K_mat/K_mat^T are precomputed
// once in bf16 (8 MiB each -> resident in the 192 MB L2 across all
// 20 iterations; per-iter traffic is 16 MiB of L2 reads).

#define KD 4
#define ND 1024
#define MD 1024
#define INV_EPS 20.0f      // 1 / 0.05
#define N_ITERS 20

typedef __bf16 bf16_t;
typedef __attribute__((ext_vector_type(8)))  __bf16 v8bf;
typedef __attribute__((ext_vector_type(16))) __bf16 v16bf;
typedef __attribute__((ext_vector_type(8)))  float  v8f;

// ---------------------------------------------------------------------------
// One-time: K_mat = exp(-costs/eps) in bf16, row-major (n-major) AND
// transposed (m-major), via LDS-tiled transpose for coalesced stores.
// ---------------------------------------------------------------------------
__global__ __launch_bounds__(256)
void expk_transpose_kernel(const float* __restrict__ costs,
                           bf16_t* __restrict__ Kb,
                           bf16_t* __restrict__ KbT) {
    __shared__ float tile[32][33];
    const int k  = blockIdx.z;
    const int m0 = blockIdx.x * 32;
    const int n0 = blockIdx.y * 32;
    const int tx = threadIdx.x;        // 0..31
    const int ty = threadIdx.y;        // 0..7
    const size_t base = (size_t)k * ND * MD;
#pragma unroll
    for (int i = 0; i < 4; ++i) {
        const int n = n0 + ty + i * 8;
        const size_t idx = base + (size_t)n * MD + (m0 + tx);
        const float e = __expf(-costs[idx] * INV_EPS);
        Kb[idx] = (bf16_t)e;
        tile[ty + i * 8][tx] = e;
    }
    __syncthreads();
#pragma unroll
    for (int i = 0; i < 4; ++i) {
        const int m = m0 + ty + i * 8;
        KbT[base + (size_t)m * ND + (n0 + tx)] = (bf16_t)tile[tx][ty + i * 8];
    }
}

// ---------------------------------------------------------------------------
// Init: v = 0  ->  exp(v) = 1
// ---------------------------------------------------------------------------
__global__ void init_state_kernel(float* __restrict__ v, bf16_t* __restrict__ ev) {
    const int i = blockIdx.x * blockDim.x + threadIdx.x;
    if (i < KD * MD) {
        v[i]  = 0.0f;
        ev[i] = (bf16_t)1.0f;
    }
}

// ---------------------------------------------------------------------------
// WMMA matvec: y = W @ x, W is 1024x1024 bf16 row-major, x is bf16[1024].
// One wave per 16-row output tile; B operand = x replicated across the 16
// columns, so D's columns are all equal to the matvec result.
//   mode 0 (row pass):  eu[row]  = dist[row] / y[row]      (u = log a - log Ku)
//   mode 1 (col pass):  col[row] = y[row]                  (colpre = K^T @ eu)
// ---------------------------------------------------------------------------
__global__ __launch_bounds__(256)
void wmma_matvec_kernel(const bf16_t* __restrict__ W,
                        const bf16_t* __restrict__ xb,
                        int mode,
                        const float* __restrict__ dist,
                        bf16_t* __restrict__ eu_out,
                        float* __restrict__ col_out) {
    const int wave = (blockIdx.x * blockDim.x + threadIdx.x) >> 5;  // 0..255
    const int lane = threadIdx.x & 31;
    const int k    = wave >> 6;          // 0..3
    const int tile = wave & 63;          // 0..63
    const int n0   = tile * 16;
    const int r    = lane & 15;
    const int hi   = lane >> 4;

    const bf16_t* Wk   = W  + (size_t)k * ND * MD;
    const bf16_t* xk   = xb + k * MD;
    const bf16_t* arow = Wk + (size_t)(n0 + r) * MD + hi * 8;

    v8f acc = {};
    for (int m0 = 0; m0 < MD; m0 += 32) {
        // A 16x32 bf16 tile, documented layout:
        //   lane r (hi=0): elems 0..7 = K 0..7,  elems 8..15 = K 16..23
        //   lane r (hi=1): elems 0..7 = K 8..15, elems 8..15 = K 24..31
        union { v16bf v; v8bf h[2]; } A;
        A.h[0] = *(const v8bf*)(arow + m0);
        A.h[1] = *(const v8bf*)(arow + m0 + 16);
        // B 32x16 bf16, columns replicated with x:
        //   same K layout as A: hi=0 -> K 0..7, 16..23; hi=1 -> K 8..15, 24..31
        union { v16bf v; v8bf h[2]; } B;
        B.h[0] = *(const v8bf*)(xk + m0 + hi * 8);
        B.h[1] = *(const v8bf*)(xk + m0 + hi * 8 + 16);
        acc = __builtin_amdgcn_wmma_f32_16x16x32_bf16(
            /*neg_a=*/false, A.v, /*neg_b=*/false, B.v,
            /*c_mod=*/(short)0, acc, /*reuse_a=*/false, /*reuse_b=*/false);
    }

    // C/D layout: VGPR j, lanes 0-15 -> M=j (N=lane); lanes 16-31 -> M=8+j.
    // Column N=0 lives in lanes 0 (rows 0..7) and 16 (rows 8..15).
    if (r == 0) {
        const int rowbase = n0 + hi * 8;
#pragma unroll
        for (int j = 0; j < 8; ++j) {
            const float y   = acc[j];
            const int   row = rowbase + j;
            if (mode == 0) {
                const float Ku = fmaxf(y, 1e-30f);
                eu_out[k * ND + row] = (bf16_t)(dist[k * ND + row] / Ku);
            } else {
                col_out[k * MD + row] = y;
            }
        }
    }
}

// ---------------------------------------------------------------------------
// Combine (one block, 1024 threads = M):
//   s[m]   = sum_k w_k * (log colpre[k,m] + v[k,m])
//   lse    = LSE_m(s);  lq = s - lse;  q = exp(lq)
//   v[k,m] = max(lq, log 1e-30) - log colpre[k,m];  ev = exp(v)
// ---------------------------------------------------------------------------
__global__ __launch_bounds__(1024)
void combine_kernel(const float* __restrict__ colpre,
                    const float* __restrict__ wts,
                    float* __restrict__ v,
                    bf16_t* __restrict__ ev,
                    float* __restrict__ q_out) {
    __shared__ float red[1024];
    const int m = threadIdx.x;

    float lcp[KD];
    float s = 0.0f;
#pragma unroll
    for (int k = 0; k < KD; ++k) {
        lcp[k] = logf(fmaxf(colpre[k * MD + m], 1e-30f));
        s += wts[k] * (lcp[k] + v[k * MD + m]);
    }

    // block LSE over m: max, then sum of exp
    red[m] = s;
    __syncthreads();
    for (int off = 512; off > 0; off >>= 1) {
        if (m < off) red[m] = fmaxf(red[m], red[m + off]);
        __syncthreads();
    }
    const float mx = red[0];
    __syncthreads();
    red[m] = expf(s - mx);
    __syncthreads();
    for (int off = 512; off > 0; off >>= 1) {
        if (m < off) red[m] += red[m + off];
        __syncthreads();
    }
    const float lse = mx + logf(red[0]);

    const float lq = s - lse;
    q_out[m] = expf(lq);                       // barycenter (last iter wins)
    const float lqc = fmaxf(lq, -69.0775527898f);  // log(1e-30) clamp (_log_safe)
#pragma unroll
    for (int k = 0; k < KD; ++k) {
        const float vn = lqc - lcp[k];
        v[k * MD + m]  = vn;
        ev[k * MD + m] = (bf16_t)expf(vn);
    }
}

// ---------------------------------------------------------------------------
// Host launcher.  Workspace (~16.8 MB):
//   Kb (8 MiB) | KbT (8 MiB) | ev bf16 | eu bf16 | v f32 | colpre f32
// ---------------------------------------------------------------------------
extern "C" void kernel_launch(void* const* d_in, const int* in_sizes, int n_in,
                              void* d_out, int out_size, void* d_ws, size_t ws_size,
                              hipStream_t stream) {
    (void)in_sizes; (void)n_in; (void)out_size; (void)ws_size;

    const float* dist  = (const float*)d_in[0];   // (K, N)
    const float* costs = (const float*)d_in[1];   // (K, N, M)
    const float* wts   = (const float*)d_in[2];   // (K,)
    float* q_out = (float*)d_out;                 // (M,) float32

    char* ws = (char*)d_ws;
    size_t off = 0;
    auto take = [&](size_t bytes) {
        void* p = ws + off;
        off = (off + bytes + 255) & ~(size_t)255;
        return p;
    };
    bf16_t* Kb  = (bf16_t*)take((size_t)KD * ND * MD * sizeof(bf16_t));
    bf16_t* KbT = (bf16_t*)take((size_t)KD * ND * MD * sizeof(bf16_t));
    bf16_t* ev  = (bf16_t*)take((size_t)KD * MD * sizeof(bf16_t));
    bf16_t* eu  = (bf16_t*)take((size_t)KD * ND * sizeof(bf16_t));
    float*  v   = (float*) take((size_t)KD * MD * sizeof(float));
    float*  col = (float*) take((size_t)KD * MD * sizeof(float));

    expk_transpose_kernel<<<dim3(MD / 32, ND / 32, KD), dim3(32, 8), 0, stream>>>(
        costs, Kb, KbT);
    init_state_kernel<<<(KD * MD + 255) / 256, 256, 0, stream>>>(v, ev);

    for (int it = 0; it < N_ITERS; ++it) {
        // row pass: Ku = K @ exp(v);  eu = a / Ku
        wmma_matvec_kernel<<<32, 256, 0, stream>>>(Kb, ev, 0, dist, eu, (float*)nullptr);
        // col pass: colpre = K^T @ eu
        wmma_matvec_kernel<<<32, 256, 0, stream>>>(KbT, eu, 1, (const float*)nullptr,
                                                   (bf16_t*)nullptr, col);
        // mix, normalize, update v / exp(v); write q
        combine_kernel<<<1, 1024, 0, stream>>>(col, wts, v, ev, q_out);
    }
}
